// AOP_NN_load_49168785604713
// MI455X (gfx1250) — matrix operand order repaired and checked by reading, not compiled, not run on hardware
//
#include <hip/hip_runtime.h>
#include <hip/hip_bf16.h>
#include <math.h>

// ---------------------------------------------------------------------------
// Tree-MLP: grouped Linear+tanh+BatchNorm(training) layers, on MI455X (gfx1250)
// bf16 WMMA (v_wmma_f32_16x16x32_bf16) grouped GEMMs; BN of layer L folded into
// layer L+1's LDS staging as per-channel scale/shift (no extra HBM pass).
// Per-layer batch stats via LDS reduction + global float atomics.
// Memory-bound (~680 MB traffic vs ~6 us of bf16 WMMA math): vectorized
// 128-bit staging loads with cross-K software prefetch, bf16 intermediates,
// coalesced 128-bit stores through an LDS output tile.
// ---------------------------------------------------------------------------

typedef __attribute__((ext_vector_type(16))) __bf16 v16bf;
typedef __attribute__((ext_vector_type(8)))  __bf16 v8bf;
typedef __attribute__((ext_vector_type(4)))  __bf16 v4bf;
typedef __attribute__((ext_vector_type(8)))  float  v8f;
typedef __attribute__((ext_vector_type(4)))  float  v4f;

#define CAT16(lo, hi) __builtin_shufflevector((lo), (hi), \
    0,1,2,3,4,5,6,7,8,9,10,11,12,13,14,15)

static constexpr int BATCH   = 16384;
static constexpr int MTILE   = 128;      // rows per block
static constexpr int KSTEP   = 32;       // bf16 WMMA K
static constexpr float BN_EPS = 1e-5f;

// BN channels per layer: 64*80, 16*80, 4*80, 64, 2
static constexpr int C1 = 5120, C2 = 1280, C3 = 320, CF = 64, CA = 2;
static constexpr int CTOT = C1 + C2 + C3 + CF + CA;   // 6786

__device__ __forceinline__ float fast_tanh(float x)
{
#if __has_builtin(__builtin_amdgcn_tanhf)
    return __builtin_amdgcn_tanhf(x);          // v_tanh_f32 (gfx1250 TRANS op)
#elif __has_builtin(__builtin_amdgcn_tanh_f32)
    return __builtin_amdgcn_tanh_f32(x);
#else
    return tanhf(x);
#endif
}

template <bool IS_F32> struct RawVec      { using type = v4bf; };
template <>            struct RawVec<true>{ using type = v4f;  };

// ---------------------------------------------------------------------------
// Grouped GEMM + tanh + stats.  in: [B, n_nodes*DI] (node j uses cols j*DI..).
// W: [n_nodes, DI, DO] fp32, bias: [n_nodes, DO].
// HAS_BN: apply per-input-channel scale/shift (previous layer's folded BN).
// out: [B, n_nodes*DO] (TOUT).  gstats: [n_nodes*DO][2] (sum,sumsq), atomics.
// ---------------------------------------------------------------------------
template <typename TIN, typename TOUT, int DI, int DO, bool HAS_BN>
__global__ __launch_bounds__(256) void node_gemm_tanh_stats(
    const TIN* __restrict__ in,
    const float* __restrict__ inscale, const float* __restrict__ inshift,
    const float* __restrict__ W, const float* __restrict__ bias,
    TOUT* __restrict__ out, float* __restrict__ gstats, int n_nodes)
{
    constexpr int NT     = (DO + 15) / 16;   // N tiles of 16
    constexpr int DO_PAD = NT * 16;
    constexpr int LDA    = KSTEP + 8;        // pads keep 16B align, avoid bank conflicts
    constexpr int LDB    = DI + 8;
    constexpr int LDO    = DO_PAD + 8;       // multiple of 8 -> 16B-aligned rows
    constexpr bool COAL  = (sizeof(TOUT) == 2) && (DO % 8 == 0);
    constexpr bool INF32 = (sizeof(TIN) == 4);
    using RawT = typename RawVec<INF32>::type;

    struct SharedAB {
        alignas(16) __bf16 A[MTILE * LDA];
        alignas(16) __bf16 B[DO_PAD * LDB];
    };
    union SharedU {
        SharedAB ab;
        alignas(16) __bf16 O[MTILE * LDO];   // output tile overlays A/B after K loop
    };
    __shared__ SharedU sh;
    __shared__ float red[DO_PAD * 2];

    const int tid     = threadIdx.x;
    const int lane    = tid & 31;
    const int wv      = tid >> 5;          // wave id = M tile (0..7)
    const int half    = lane >> 4;         // 0/1
    const int l16     = lane & 15;
    const int rowbase = blockIdx.x * MTILE;
    const int node    = blockIdx.y;
    const int in_stride  = n_nodes * DI;
    const int out_stride = n_nodes * DO;

    // ---- stage full node weights into LDS, transposed to [N][K] (bf16) ----
    for (int idx = tid; idx < DO_PAD * DI; idx += 256) {
        int nn = idx % DO_PAD;             // consecutive threads -> consecutive N: coalesced
        int kk = idx / DO_PAD;
        float w = (nn < DO) ? W[((size_t)node * DI + kk) * DO + nn] : 0.0f;
        sh.ab.B[nn * LDB + kk] = (__bf16)w;
    }
    for (int i = tid; i < DO_PAD * 2; i += 256) red[i] = 0.0f;

    v8f acc[NT];
    const v8f zacc = {0.f, 0.f, 0.f, 0.f, 0.f, 0.f, 0.f, 0.f};
#pragma unroll
    for (int i = 0; i < NT; ++i) acc[i] = zacc;

    const int tx = tid & 7;                // col group of 4 elems (32 cols / 8)
    const int ty = tid >> 3;               // 0..31 (4 row passes -> 128 rows)

    // prefetch registers: 4 row-chunks of the A tile + BN scale/shift vectors
    RawT raw[4];
    v4f  sc, sf;
    auto load_tile = [&](int k0) {
        const size_t cbase = (size_t)node * DI + k0 + tx * 4;
#pragma unroll
        for (int j = 0; j < 4; ++j) {      // all 4 loads live -> issued back-to-back
            raw[j] = *(const RawT*)(in + (size_t)(rowbase + ty + 32 * j) * in_stride
                                       + cbase);
        }
        if constexpr (HAS_BN) {
            sc = *(const v4f*)(inscale + cbase);   // row-invariant 16B vectors
            sf = *(const v4f*)(inshift + cbase);
        }
    };

    load_tile(0);                          // prologue prefetch
    __syncthreads();                       // covers weight staging + red init

    for (int k0 = 0; k0 < DI; k0 += KSTEP) {
        // ---- stage prefetched A tile [128 x 32]: BN fold + bf16 pack ----
#pragma unroll
        for (int j = 0; j < 4; ++j) {
            v4f vv;
            if constexpr (INF32) {
                vv = raw[j];
            } else {
#pragma unroll
                for (int e = 0; e < 4; ++e) vv[e] = (float)raw[j][e];
            }
            if constexpr (HAS_BN) vv = sc * vv + sf;
            v4bf pk;
#pragma unroll
            for (int e = 0; e < 4; ++e) pk[e] = (__bf16)vv[e];
            *(v4bf*)(&sh.ab.A[(ty + 32 * j) * LDA + tx * 4]) = pk;  // ds_store_b64
        }
        __syncthreads();

        // issue next K-step's global loads now: latency hides under WMMA work
        if (k0 + KSTEP < DI) load_tile(k0 + KSTEP);

        // ---- A fragment (ISA 16-bit A 16x32 layout): two b128 LDS loads ----
        const __bf16* arow = &sh.ab.A[(wv * 16 + l16) * LDA];
        v8bf a0 = *(const v8bf*)(arow + half * 8);        // K = half*8 + 0..7
        v8bf a1 = *(const v8bf*)(arow + 16 + half * 8);   // K = 16 + half*8 + 0..7
        v16bf afrag = CAT16(a0, a1);

#pragma unroll
        for (int nt = 0; nt < NT; ++nt) {
            // B fragment (32x16 K-major/col): K = half*8 + 0..7, then 16 + half*8 + 0..7
            const __bf16* brow = &sh.ab.B[(nt * 16 + l16) * LDB + k0 + half * 8];
            v8bf b0 = *(const v8bf*)(brow);
            v8bf b1 = *(const v8bf*)(brow + 16);
            v16bf bfrag = CAT16(b0, b1);
            acc[nt] = __builtin_amdgcn_wmma_f32_16x16x32_bf16(
                false, afrag, false, bfrag, (short)0, acc[nt], false, false);
        }
        __syncthreads();   // frag reads done before next staging overwrites A
    }

    // ---- epilogue: bias + tanh + stats; outputs via LDS tile (coalesced) ----
#pragma unroll
    for (int nt = 0; nt < NT; ++nt) {
        const int col = nt * 16 + l16;
        if (col < DO) {
            const float bb = bias[node * DO + col];
            float s = 0.f, sq = 0.f;
#pragma unroll
            for (int v = 0; v < 8; ++v) {                 // C/D layout: row = v + 8*half
                const float t = fast_tanh(acc[nt][v] + bb);
                const int rloc = wv * 16 + half * 8 + v;
                if constexpr (COAL) {
                    sh.O[rloc * LDO + col] = (__bf16)t;   // overlays A/B (post-barrier)
                } else {
                    out[(size_t)(rowbase + rloc) * out_stride
                        + (size_t)node * DO + col] = (TOUT)t;
                }
                s += t; sq += t * t;
            }
            atomicAdd(&red[col * 2 + 0], s);
            atomicAdd(&red[col * 2 + 1], sq);
        }
    }
    __syncthreads();

    if constexpr (COAL) {
        // 16B per thread over contiguous channels -> global_store_b128
        for (int idx = tid * 8; idx < MTILE * DO; idx += 256 * 8) {
            const int row = idx / DO;
            const int col = idx % DO;                     // DO % 8 == 0: no row crossing
            v8bf vv = *(const v8bf*)(&sh.O[row * LDO + col]);
            *(v8bf*)((__bf16*)out + (size_t)(rowbase + row) * out_stride
                     + (size_t)node * DO + col) = vv;
        }
    }

    for (int i = tid; i < DO * 2; i += 256)
        atomicAdd(&gstats[(size_t)node * DO * 2 + i], red[i]);
}

// stats -> fused BN scale/shift:  y = scale*h + shift
__global__ void bn_params_kernel(const float* __restrict__ stats,
                                 const float* __restrict__ g,
                                 const float* __restrict__ c,
                                 float* __restrict__ scale,
                                 float* __restrict__ shift,
                                 int C, float invB)
{
    int i = blockIdx.x * 256 + threadIdx.x;
    if (i < C) {
        float mean = stats[2 * i] * invB;
        float var  = stats[2 * i + 1] * invB - mean * mean;
        float rsig = rsqrtf(var + BN_EPS);
        scale[i] = g[i] * rsig;
        shift[i] = c[i] - g[i] * mean * rsig;
    }
}

__global__ void zero_f32_kernel(float* __restrict__ p, int n)
{
    int i = blockIdx.x * 256 + threadIdx.x;
    if (i < n) p[i] = 0.0f;
}

// final output: apply aux-head BN
__global__ void apply_bn_out_kernel(const float* __restrict__ h,
                                    const float* __restrict__ scale,
                                    const float* __restrict__ shift,
                                    float* __restrict__ out, int n, int C)
{
    int i = blockIdx.x * 256 + threadIdx.x;
    if (i < n) {
        int c = i % C;
        out[i] = scale[c] * h[i] + shift[c];
    }
}

// ---------------------------------------------------------------------------
extern "C" void kernel_launch(void* const* d_in, const int* in_sizes, int n_in,
                              void* d_out, int out_size, void* d_ws, size_t ws_size,
                              hipStream_t stream)
{
    const float* x  = (const float*)d_in[0];
    const float* W1 = (const float*)d_in[1];
    const float* b1 = (const float*)d_in[2];
    const float* g1 = (const float*)d_in[3];
    const float* c1 = (const float*)d_in[4];
    const float* W2 = (const float*)d_in[5];
    const float* b2 = (const float*)d_in[6];
    const float* g2 = (const float*)d_in[7];
    const float* c2 = (const float*)d_in[8];
    const float* W3 = (const float*)d_in[9];
    const float* b3 = (const float*)d_in[10];
    const float* g3 = (const float*)d_in[11];
    const float* c3 = (const float*)d_in[12];
    const float* Wf = (const float*)d_in[13];
    const float* bf = (const float*)d_in[14];
    const float* gf = (const float*)d_in[15];
    const float* cf = (const float*)d_in[16];
    const float* Wa = (const float*)d_in[17];
    const float* ba = (const float*)d_in[18];
    const float* ga = (const float*)d_in[19];
    const float* ca = (const float*)d_in[20];

    char* ws = (char*)d_ws;
    size_t off = 0;
    auto take = [&](size_t bytes) -> char* {
        char* p = ws + off;
        off = (off + bytes + 255) & ~(size_t)255;
        return p;
    };
    __bf16* h1 = (__bf16*)take((size_t)BATCH * C1 * sizeof(__bf16));  // [B,5120]
    __bf16* h2 = (__bf16*)take((size_t)BATCH * C2 * sizeof(__bf16));  // [B,1280]
    __bf16* h3 = (__bf16*)take((size_t)BATCH * C3 * sizeof(__bf16));  // [B,320]
    __bf16* hf = (__bf16*)take((size_t)BATCH * CF * sizeof(__bf16));  // [B,64]
    float*  ha = (float*)take((size_t)BATCH * CA * sizeof(float));    // [B,2]
    float*  stats = (float*)take((size_t)CTOT * 2 * sizeof(float));
    float*  scale = (float*)take((size_t)CTOT * sizeof(float));
    float*  shift = (float*)take((size_t)CTOT * sizeof(float));

    const float invB = 1.0f / (float)BATCH;
    const dim3 blk(256);
    const int mb = BATCH / MTILE;   // 128 batch tiles

    zero_f32_kernel<<<(CTOT * 2 + 255) / 256, blk, 0, stream>>>(stats, CTOT * 2);

    // layer 1: 64 nodes, [B,64] x [64,80]
    node_gemm_tanh_stats<float, __bf16, 64, 80, false><<<dim3(mb, 64), blk, 0, stream>>>(
        x, nullptr, nullptr, W1, b1, h1, stats, 64);
    bn_params_kernel<<<(C1 + 255) / 256, blk, 0, stream>>>(
        stats, g1, c1, scale, shift, C1, invB);

    // layer 2: 16 nodes, [B,320] x [320,80]
    node_gemm_tanh_stats<__bf16, __bf16, 320, 80, true><<<dim3(mb, 16), blk, 0, stream>>>(
        h1, scale, shift, W2, b2, h2, stats + 2 * C1, 16);
    bn_params_kernel<<<(C2 + 255) / 256, blk, 0, stream>>>(
        stats + 2 * C1, g2, c2, scale + C1, shift + C1, C2, invB);

    // layer 3: 4 nodes, [B,320] x [320,80]
    node_gemm_tanh_stats<__bf16, __bf16, 320, 80, true><<<dim3(mb, 4), blk, 0, stream>>>(
        h2, scale + C1, shift + C1, W3, b3, h3, stats + 2 * (C1 + C2), 4);
    bn_params_kernel<<<(C3 + 255) / 256, blk, 0, stream>>>(
        stats + 2 * (C1 + C2), g3, c3, scale + C1 + C2, shift + C1 + C2, C3, invB);

    // final node: [B,320] x [320,64]
    node_gemm_tanh_stats<__bf16, __bf16, 320, 64, true><<<dim3(mb, 1), blk, 0, stream>>>(
        h3, scale + C1 + C2, shift + C1 + C2, Wf, bf, hf,
        stats + 2 * (C1 + C2 + C3), 1);
    bn_params_kernel<<<1, blk, 0, stream>>>(
        stats + 2 * (C1 + C2 + C3), gf, cf,
        scale + C1 + C2 + C3, shift + C1 + C2 + C3, CF, invB);

    // aux head: [B,64] x [64,2] (N padded to 16 inside kernel; direct f32 stores)
    node_gemm_tanh_stats<__bf16, float, 64, 2, true><<<dim3(mb, 1), blk, 0, stream>>>(
        hf, scale + C1 + C2 + C3, shift + C1 + C2 + C3, Wa, ba, ha,
        stats + 2 * (C1 + C2 + C3 + CF), 1);
    bn_params_kernel<<<1, blk, 0, stream>>>(
        stats + 2 * (C1 + C2 + C3 + CF), ga, ca,
        scale + C1 + C2 + C3 + CF, shift + C1 + C2 + C3 + CF, CA, invB);

    // apply final BN -> d_out [B,2] fp32
    apply_bn_out_kernel<<<(BATCH * CA + 255) / 256, blk, 0, stream>>>(
        ha, scale + C1 + C2 + C3 + CF, shift + C1 + C2 + C3 + CF,
        (float*)d_out, BATCH * CA, CA);
}
